// GNN_13176959664143
// MI455X (gfx1250) — matrix operand order repaired and checked by reading, not compiled, not run on hardware
//
#include <hip/hip_runtime.h>
#include <cstdint>

// ---------------------------------------------------------------------------
// GCN (2-layer) for MI455X / gfx1250.
//   t = X @ W via v_wmma_f32_16x16x32_f16 (B pre-packed to fragment layout),
//   GEMM epilogue writes both t and agg = t*dinv^2 + bias (self-loop + bias),
//   edge scatter: agg[dst] += t[src]*dinv[src]*dinv[dst] via no-return
//   global_atomic_add_f32, ReLU fused into GEMM-2's A-operand conversion.
// Edge aggregation dominates (~2.5 GB gather+atomic traffic, L2-resident:
// per-layer feature arrays are 51/26 MB << 192 MB L2). Edge index reads are
// non-temporal so they don't evict the feature arrays.
// ---------------------------------------------------------------------------

typedef __attribute__((ext_vector_type(16))) _Float16 v16h;
typedef __attribute__((ext_vector_type(8)))  float    v8f;

#define DIN 128

__device__ __forceinline__ long long ntload_i64(const long long* p) {
  return __builtin_nontemporal_load(p);
}

// ---- degree / normalization -----------------------------------------------
__global__ void k_fill1(float* __restrict__ p, int n) {
  int i = blockIdx.x * blockDim.x + threadIdx.x;
  if (i < n) p[i] = 1.0f;               // self-loop contributes 1 to degree
}

__global__ void k_count_deg(const long long* __restrict__ dst,
                            float* __restrict__ deg, int E) {
  int i = blockIdx.x * blockDim.x + threadIdx.x;
  if (i < E) unsafeAtomicAdd(&deg[(int)ntload_i64(&dst[i])], 1.0f);
}

__global__ void k_rsqrt(float* __restrict__ p, int n) {
  int i = blockIdx.x * blockDim.x + threadIdx.x;
  if (i < n) p[i] = rsqrtf(p[i]);       // deg >= 1 always
}

// ---- pack W[128,DOUT] f32 -> WMMA B-fragment layout, f16 -------------------
// Fragment (ktile, coltile): lane holds col N = lane%16, K = ktile*32 +
// (lane/16)*16 + [0..15]; stored as 16 contiguous f16 per lane.
template <int DOUT>
__global__ void k_pack_w(const float* __restrict__ W, _Float16* __restrict__ Wp) {
  constexpr int CT = DOUT / 16;                 // col tiles
  const int idx = blockIdx.x * blockDim.x + threadIdx.x;
  if (idx >= 4 * CT * 32) return;               // 4 K-tiles of 32
  const int lane = idx & 31;
  const int ct   = (idx >> 5) % CT;
  const int kt   = idx / (CT * 32);
  const int col  = ct * 16 + (lane & 15);
  const int kb   = kt * 32 + (lane >> 4) * 16;
  _Float16* o = Wp + (size_t)idx * 16;          // idx == (kt*CT+ct)*32+lane
#pragma unroll
  for (int i = 0; i < 16; ++i)
    o[i] = (_Float16)W[(size_t)(kb + i) * DOUT + col];
}

// ---- fused transform: T = X @ W ; AGG = T*dinv^2 + bias --------------------
// One wave32 per 16x16 tile, K=128 in four 16x16x32 f16 WMMAs.
// RELUA: apply ReLU to X while converting to f16 (layer-2 consumes relu(h)).
template <int DOUT, bool RELUA>
__global__ void k_gemm_fused(const float* __restrict__ X,
                             const _Float16* __restrict__ Wp,
                             float* __restrict__ T,
                             float* __restrict__ AGG,
                             const float* __restrict__ dinv,
                             const float* __restrict__ bias, int nrows) {
  constexpr int CT = DOUT / 16;
  const int lane = threadIdx.x & 31;
  const int wave = threadIdx.x >> 5;      // column tile within block
  const int l16  = lane & 15;
  const int half = lane >> 4;
  const int rowBase = blockIdx.x * 16;
  const int colBase = wave * 16;

  int arow = rowBase + l16;               // clamp keeps EXEC all-ones
  if (arow >= nrows) arow = nrows - 1;

  const v16h* Wf = (const v16h*)Wp;
  v8f acc = {};
#pragma unroll
  for (int kt = 0; kt < DIN / 32; ++kt) {
    v16h a;
    // A: lane holds row M=l16; VGPR i -> K = i*2 (+16 if half)
    const float* xr = X + (size_t)arow * DIN + kt * 32 + half * 16;
#pragma unroll
    for (int i = 0; i < 8; ++i) {
      const int k = i << 1;
      float v0 = xr[k], v1 = xr[k + 1];
      if (RELUA) { v0 = fmaxf(v0, 0.0f); v1 = fmaxf(v1, 0.0f); }
      a[2 * i]     = (_Float16)v0;
      a[2 * i + 1] = (_Float16)v1;
    }
    // B: one 32-byte vector load from the pre-packed fragment
    const v16h b = Wf[(size_t)(kt * CT + wave) * 32 + lane];
    acc = __builtin_amdgcn_wmma_f32_16x16x32_f16(
        /*neg_a=*/false, a, /*neg_b=*/false, b,
        /*c_mod=*/(short)0, acc, /*reuse_a=*/false, /*reuse_b=*/false);
  }
  // Epilogue: C/D VGPR r -> row rowBase + half*8 + r, col colBase + l16.
  const float bv = bias[colBase + l16];
#pragma unroll
  for (int r = 0; r < 8; ++r) {
    const int row = rowBase + half * 8 + r;
    if (row < nrows) {
      const float s = dinv[row];
      const size_t o = (size_t)row * DOUT + colBase + l16;
      T[o]   = acc[r];
      AGG[o] = acc[r] * (s * s) + bv;   // self-loop + bias (pre-scatter init)
    }
  }
}

// ---- edge scatter: one wave32 per edge, float4/float2 per lane ------------
template <int D>
__global__ void k_scatter(const long long* __restrict__ src,
                          const long long* __restrict__ dst,
                          const float* __restrict__ dinv,
                          const float* __restrict__ T,
                          float* __restrict__ AGG, int E) {
  constexpr int PE = D / 32;                       // elems per lane (4 or 2)
  const int lane = threadIdx.x & 31;
  const int e = (blockIdx.x * blockDim.x + threadIdx.x) >> 5;
  if (e >= E) return;
  const int s = (int)ntload_i64(&src[e]);
  const int d = (int)ntload_i64(&dst[e]);
  const float norm = dinv[s] * dinv[d];
  const float* ts = T + (size_t)s * D + lane * PE; // coalesced 512B/256B gather
  float* ad = AGG + (size_t)d * D + lane * PE;
  if constexpr (PE == 4) {
    const float4 v = *(const float4*)ts;
    unsafeAtomicAdd(ad + 0, v.x * norm);
    unsafeAtomicAdd(ad + 1, v.y * norm);
    unsafeAtomicAdd(ad + 2, v.z * norm);
    unsafeAtomicAdd(ad + 3, v.w * norm);
  } else {
    const float2 v = *(const float2*)ts;
    unsafeAtomicAdd(ad + 0, v.x * norm);
    unsafeAtomicAdd(ad + 1, v.y * norm);
  }
}

// ---------------------------------------------------------------------------
extern "C" void kernel_launch(void* const* d_in, const int* in_sizes, int n_in,
                              void* d_out, int out_size, void* d_ws,
                              size_t ws_size, hipStream_t stream) {
  const float*     x    = (const float*)d_in[0];
  const long long* edge = (const long long*)d_in[1];   // int64 per reference
  const float*     W1   = (const float*)d_in[2];
  const float*     b1   = (const float*)d_in[3];
  const float*     W2   = (const float*)d_in[4];
  const float*     b2   = (const float*)d_in[5];

  const int N = in_sizes[0] / DIN;     // 100000
  const int E = in_sizes[1] / 2;       // 1600000
  const long long* srcI = edge;        // edge_index[0]
  const long long* dstI = edge + E;    // edge_index[1]

  // ws: dinv | t (N*128 f32) | agg (N*128 f32) | Wp1 (32KB) | Wp2 (16KB)
  char*  ws   = (char*)d_ws;
  float* dinv = (float*)ws;
  size_t off  = (((size_t)N * sizeof(float)) + 511) & ~(size_t)511;
  float* t    = (float*)(ws + off);
  size_t fsz  = (size_t)N * DIN * sizeof(float);
  float* agg  = (float*)(ws + off + fsz);
  _Float16* Wp1 = (_Float16*)(ws + off + 2 * fsz);
  _Float16* Wp2 = Wp1 + 4 * 8 * 32 * 16;   // after 32KB of Wp1
  float* out  = (float*)d_out;

  // degrees (including self-loop) -> dinv = rsqrt(deg)
  k_fill1<<<(N + 255) / 256, 256, 0, stream>>>(dinv, N);
  k_count_deg<<<(E + 255) / 256, 256, 0, stream>>>(dstI, dinv, E);
  k_rsqrt<<<(N + 255) / 256, 256, 0, stream>>>(dinv, N);

  // pack weights into WMMA B-fragment layout (f16)
  k_pack_w<128><<<(4 * 8 * 32 + 255) / 256, 256, 0, stream>>>(W1, Wp1);
  k_pack_w<64><<<(4 * 4 * 32 + 255) / 256, 256, 0, stream>>>(W2, Wp2);

  // ---- layer 1: 128 -> 128 (relu deferred into layer-2 A conversion) ----
  k_gemm_fused<128, false><<<(N + 15) / 16, 256, 0, stream>>>(
      x, Wp1, t, agg, dinv, b1, N);
  k_scatter<128><<<(E + 7) / 8, 256, 0, stream>>>(srcI, dstI, dinv, t, agg, E);

  // ---- layer 2: 128 -> 64 ----
  k_gemm_fused<64, true><<<(N + 15) / 16, 128, 0, stream>>>(
      agg, Wp2, t, out, dinv, b2, N);
  k_scatter<64><<<(E + 7) / 8, 256, 0, stream>>>(srcI, dstI, dinv, t, out, E);
}
